// TimeAttention_18829136626260
// MI455X (gfx1250) — matrix operand order repaired and checked by reading, not compiled, not run on hardware
//
#include <hip/hip_runtime.h>

// ---------------------------------------------------------------------------
// TimeAttention for MI455X (gfx1250, wave32, WMMA bf16 16x16x32, f32 accum)
//
// Reference math (B=64, N=16 -> NV=17, T=128, H=256, f32):
//   kv      = concat(node, neigh)                       [B,17,T,H]
//   S       = (node @ Wq^T @ Wb @ Wk) @ kv^T + b        [B,17,T,T]
//   A       = softmax(S, axis=-1)
//   outputs = (A @ kv) @ Wv^T                           [B,17,T,H]
// Folding Wq^T@Wb@Wk into one HxH matrix and reassociating A@value removes
// the key/value intermediates entirely (memory-bound problem: ~360MB/call).
// Wv^T (128 KB) is prefetched into LDS with GLOBAL_LOAD_ASYNC_TO_LDS_B128
// (ASYNCcnt) at kernel entry and drained with s_wait_asynccnt only right
// before the P @ Wv^T phase, hiding the fetch behind S/softmax/A@kv compute.
// ---------------------------------------------------------------------------

#define B_  64
#define N_  16
#define NV_ 17
#define T_  128
#define H_  256

#if defined(__HIP_DEVICE_COMPILE__) && defined(__has_builtin)
#if __has_builtin(__builtin_amdgcn_global_load_async_to_lds_b128)
#define USE_ASYNC_WVT 1
#endif
#endif
#ifndef USE_ASYNC_WVT
#define USE_ASYNC_WVT 0
#endif

typedef __attribute__((ext_vector_type(16))) __bf16 v16bf;
typedef __attribute__((ext_vector_type(8)))  float  v8f;
typedef __attribute__((ext_vector_type(4)))  int    v4i;

#if USE_ASYNC_WVT
// address-space-qualified pointee types for the async-to-LDS builtin
typedef __attribute__((address_space(1))) v4i g_v4i;   // global
typedef __attribute__((address_space(3))) v4i l_v4i;   // LDS
#endif

union Frag {
  v16bf v;
  unsigned short u[16];
  unsigned int   d[8];
};

__device__ __forceinline__ unsigned short f2bf(float x) {
  unsigned int u = __float_as_uint(x);
  u += 0x7FFFu + ((u >> 16) & 1u);      // round-to-nearest-even
  return (unsigned short)(u >> 16);
}
__device__ __forceinline__ float bf2f(unsigned short h) {
  return __uint_as_float(((unsigned int)h) << 16);
}
__device__ __forceinline__ v8f vzero8() {
  v8f z;
#pragma unroll
  for (int i = 0; i < 8; ++i) z[i] = 0.0f;
  return z;
}

// A-fragment (16x32 bf16, MxK). Contiguous k order (same as B frags): all lanes M=lane%16;
// element i -> K = half*16 + i.
__device__ __forceinline__ v16bf frag_a(const unsigned short* src, int ld,
                                        int row0, int k0, int lane) {
  const int half = (lane >> 4) & 1, m = lane & 15;
  Frag f;
  const unsigned short* p = src + (size_t)(row0 + m) * ld + k0 + half * 16;
#pragma unroll
  for (int q = 0; q < 8; ++q) f.d[q] = *(const unsigned int*)(p + 2 * q);
  return f.v;
}

// B-fragment (32x16 bf16, KxN) where B[k][n] = src[n0+n][k0+k] (i.e. src^T),
// contiguous along k. Order used here: N = lane%16, element i -> K = half*16 + i.
__device__ __forceinline__ v16bf frag_bT(const unsigned short* src, int ld,
                                         int n0, int k0, int lane) {
  const int half = (lane >> 4) & 1, m = lane & 15;
  Frag f;
  const unsigned short* p = src + (size_t)(n0 + m) * ld + k0 + half * 16;
#pragma unroll
  for (int q = 0; q < 8; ++q) f.d[q] = *(const unsigned int*)(p + 2 * q);
  return f.v;
}

// B-fragment (32x16 bf16, KxN) where B[k][n] = src[k0+k][n0+n] (row-major),
// strided along k.
__device__ __forceinline__ v16bf frag_b(const unsigned short* src, int ld,
                                        int k0, int n0, int lane) {
  const int half = (lane >> 4) & 1, m = lane & 15;
  Frag f;
  const unsigned short* p = src + (size_t)(k0 + half * 16) * ld + n0 + m;
#pragma unroll
  for (int i = 0; i < 16; ++i) f.u[i] = p[(size_t)i * ld];
  return f.v;
}

// ---------------------------------------------------------------------------
// Kernel 1: Wqb = Wq^T @ Wb   (f32, 256x256, negligible cost)
__global__ void k_wqb(const float* __restrict__ Wq, const float* __restrict__ Wb,
                      float* __restrict__ Wqb) {
  const int h = blockIdx.x, j = threadIdx.x;
  float s = 0.0f;
  for (int m = 0; m < H_; ++m) s += Wq[m * H_ + h] * Wb[m * H_ + j];
  Wqb[h * H_ + j] = s;
}

// Kernel 2: Wg = Wqb @ Wk (bf16) ;  WvT[h][j] = Wv[j][h] (bf16)
__global__ void k_wg_wvt(const float* __restrict__ Wqb, const float* __restrict__ Wk,
                         const float* __restrict__ Wv,
                         unsigned short* __restrict__ Wg,
                         unsigned short* __restrict__ WvT) {
  const int h = blockIdx.x, t = threadIdx.x;
  float s = 0.0f;
  for (int j = 0; j < H_; ++j) s += Wqb[h * H_ + j] * Wk[j * H_ + t];
  Wg[h * H_ + t]  = f2bf(s);
  WvT[h * H_ + t] = f2bf(Wv[t * H_ + h]);
}

// Kernel 3: G = node @ Wg  ->  bf16 [B*T, H]  (1 GFLOP, negligible)
__global__ void k_g(const float* __restrict__ node, const unsigned short* __restrict__ Wg,
                    unsigned short* __restrict__ G) {
  const int bt = blockIdx.x, h = threadIdx.x;
  const float* row = node + (size_t)bt * H_;
  float s = 0.0f;
  for (int k = 0; k < H_; ++k) s += row[k] * bf2f(Wg[k * H_ + h]);
  G[(size_t)bt * H_ + h] = f2bf(s);
}

// ---------------------------------------------------------------------------
// Kernel 4: fused attention. One workgroup per (b, n) slot; 8 waves, each
// owning a 16-row stripe of T. Dynamic LDS partition (288 KB of 320 KB WGP):
//   s_kv  [128][256] bf16  (64 KB)   kv tile, converted f32->bf16 on load
//   s_wvt [256][256] bf16 (128 KB)   Wv^T, async-prefetched once per WG
//   s_a   [128][128] bf16  (32 KB)   softmaxed A (staging for P = A@kv)
//   s_p   [128][256] bf16  (64 KB)   P = A@kv   (staging for P@Wv^T)
__global__ void k_attn(const float* __restrict__ node, const float* __restrict__ neigh,
                       const float* __restrict__ bvec,
                       const unsigned short* __restrict__ G,
                       const unsigned short* __restrict__ WvT,
                       float* __restrict__ outP, float* __restrict__ outA) {
  const int bn = blockIdx.x;
  const int b  = bn / NV_;
  const int n  = bn % NV_;
  const int tid  = threadIdx.x;
  const int wave = tid >> 5;
  const int lane = tid & 31;
  const int half = (lane >> 4) & 1;
  const int mcol = lane & 15;
  const int t0   = wave * 16;           // this wave's row stripe in T

  extern __shared__ char smem[];
  unsigned short* s_kv  = (unsigned short*)smem;          // 128*256
  unsigned short* s_wvt = s_kv + T_ * H_;                 // 256*256
  unsigned short* s_a   = s_wvt + H_ * H_;                // 128*128
  unsigned short* s_p   = s_a + T_ * T_;                  // 128*256

  // ---- Wv^T prefetch: async global->LDS, drained only before phase 3 -----
#if USE_ASYNC_WVT
  for (int i = tid; i < (H_ * H_) / 8; i += 256) {
    __builtin_amdgcn_global_load_async_to_lds_b128(
        (g_v4i*)(WvT + (size_t)i * 8),
        (l_v4i*)(s_wvt + (size_t)i * 8),
        0, 0);
  }
#else
  for (int i = tid; i < (H_ * H_) / 8; i += 256)
    ((uint4*)s_wvt)[i] = ((const uint4*)WvT)[i];
#endif

  // ---- kv tile load (needed immediately by phase 1) ----------------------
  const float* kv_src = (n == 0) ? (node + (size_t)b * T_ * H_)
                                 : (neigh + ((size_t)b * N_ + (n - 1)) * T_ * H_);
  for (int i = tid; i < (T_ * H_) / 4; i += 256) {
    const float4 v = ((const float4*)kv_src)[i];
    unsigned int lo = (unsigned int)f2bf(v.x) | ((unsigned int)f2bf(v.y) << 16);
    unsigned int hi = (unsigned int)f2bf(v.z) | ((unsigned int)f2bf(v.w) << 16);
    ((unsigned int*)s_kv)[2 * i]     = lo;
    ((unsigned int*)s_kv)[2 * i + 1] = hi;
  }
  __syncthreads();   // s_kv visible to all waves (s_wvt not needed yet)

  const unsigned short* Gb = G + (size_t)b * T_ * H_;

  // ---- phase 1: S = G @ kv^T (+b), softmax -> A --------------------------
  v8f accS[8];
#pragma unroll
  for (int j = 0; j < 8; ++j) accS[j] = vzero8();

  for (int k0 = 0; k0 < H_; k0 += 32) {
    const v16bf a = frag_a(Gb, H_, t0, k0, lane);
#pragma unroll
    for (int j = 0; j < 8; ++j) {
      const v16bf bb = frag_bT(s_kv, H_, j * 16, k0, lane);
      accS[j] = __builtin_amdgcn_wmma_f32_16x16x32_bf16(
          false, a, false, bb, (short)0, accS[j], false, false);
    }
  }

  // bias + row max (rows live in VGPR r across the 16 lanes of each half)
  float rowm[8];
#pragma unroll
  for (int r = 0; r < 8; ++r) rowm[r] = -3.4e38f;
#pragma unroll
  for (int j = 0; j < 8; ++j) {
    const float bj = bvec[j * 16 + mcol];
#pragma unroll
    for (int r = 0; r < 8; ++r) {
      accS[j][r] += bj;
      rowm[r] = fmaxf(rowm[r], accS[j][r]);
    }
  }
#pragma unroll
  for (int r = 0; r < 8; ++r)
    for (int mask = 1; mask < 16; mask <<= 1)
      rowm[r] = fmaxf(rowm[r], __shfl_xor(rowm[r], mask, 32));

  float rows[8];
#pragma unroll
  for (int r = 0; r < 8; ++r) rows[r] = 0.0f;
#pragma unroll
  for (int j = 0; j < 8; ++j)
#pragma unroll
    for (int r = 0; r < 8; ++r) {
      accS[j][r] = __expf(accS[j][r] - rowm[r]);
      rows[r] += accS[j][r];
    }
#pragma unroll
  for (int r = 0; r < 8; ++r) {
    for (int mask = 1; mask < 16; mask <<= 1)
      rows[r] += __shfl_xor(rows[r], mask, 32);
    rows[r] = 1.0f / rows[r];
  }

  // normalize, emit A (f32 -> global out, bf16 -> LDS for P = A@kv)
  float* outAb = outA + ((size_t)bn * T_) * T_;
#pragma unroll
  for (int j = 0; j < 8; ++j)
#pragma unroll
    for (int r = 0; r < 8; ++r) {
      const float av = accS[j][r] * rows[r];
      const int row = t0 + r + 8 * half;
      const int col = j * 16 + mcol;
      outAb[(size_t)row * T_ + col] = av;
      s_a[row * T_ + col] = f2bf(av);
    }
  __syncthreads();

  // ---- phase 2: P = A @ kv  (bf16 into LDS) ------------------------------
  v16bf afr[4];
#pragma unroll
  for (int kk = 0; kk < 4; ++kk) afr[kk] = frag_a(s_a, T_, t0, kk * 32, lane);

  for (int ht = 0; ht < 16; ++ht) {
    v8f acc = vzero8();
#pragma unroll
    for (int kk = 0; kk < 4; ++kk) {
      const v16bf bb = frag_b(s_kv, H_, kk * 32, ht * 16, lane);
      acc = __builtin_amdgcn_wmma_f32_16x16x32_bf16(
          false, afr[kk], false, bb, (short)0, acc, false, false);
    }
#pragma unroll
    for (int r = 0; r < 8; ++r)
      s_p[(t0 + r + 8 * half) * H_ + ht * 16 + mcol] = f2bf(acc[r]);
  }

  // ---- drain the Wv^T async prefetch, then sync before phase 3 -----------
#if USE_ASYNC_WVT
#if __has_builtin(__builtin_amdgcn_s_wait_asynccnt)
  __builtin_amdgcn_s_wait_asynccnt(0);
#else
  asm volatile("s_wait_asynccnt 0" ::: "memory");
#endif
#endif
  __syncthreads();

  // ---- phase 3: outputs = P @ Wv^T ---------------------------------------
  v16bf pfr[8];
#pragma unroll
  for (int kk = 0; kk < 8; ++kk) pfr[kk] = frag_a(s_p, H_, t0, kk * 32, lane);

  float* outPb = outP + ((size_t)bn * T_) * H_;
  for (int jt = 0; jt < 16; ++jt) {
    v8f acc = vzero8();
#pragma unroll
    for (int kk = 0; kk < 8; ++kk) {
      const v16bf bb = frag_b(s_wvt, H_, kk * 32, jt * 16, lane);
      acc = __builtin_amdgcn_wmma_f32_16x16x32_bf16(
          false, pfr[kk], false, bb, (short)0, acc, false, false);
    }
#pragma unroll
    for (int r = 0; r < 8; ++r)
      outPb[(size_t)(t0 + r + 8 * half) * H_ + jt * 16 + mcol] = acc[r];
  }
}

// ---------------------------------------------------------------------------
extern "C" void kernel_launch(void* const* d_in, const int* in_sizes, int n_in,
                              void* d_out, int out_size, void* d_ws, size_t ws_size,
                              hipStream_t stream) {
  const float* node  = (const float*)d_in[0];   // [B,T,H]
  const float* neigh = (const float*)d_in[1];   // [B,N,T,H]
  const float* Wq    = (const float*)d_in[3];   // [H,H]
  const float* Wk    = (const float*)d_in[4];
  const float* Wv    = (const float*)d_in[5];
  const float* Wb    = (const float*)d_in[6];
  const float* bvec  = (const float*)d_in[7];   // [T]

  char* ws = (char*)d_ws;
  float*          Wqb = (float*)ws;                              // 256 KB
  unsigned short* Wg  = (unsigned short*)(ws + 262144);          // 128 KB
  unsigned short* WvT = (unsigned short*)(ws + 262144 + 131072); // 128 KB
  unsigned short* G   = (unsigned short*)(ws + 524288);          // 4 MB

  float* outP = (float*)d_out;                          // [B,17,T,H]
  float* outA = outP + (size_t)B_ * NV_ * T_ * H_;      // [B,17,T,T]

  k_wqb   <<<H_, H_, 0, stream>>>(Wq, Wb, Wqb);
  k_wg_wvt<<<H_, H_, 0, stream>>>(Wqb, Wk, Wv, Wg, WvT);
  k_g     <<<B_ * T_, H_, 0, stream>>>(node, Wg, G);

  const int smem = (T_ * H_ + H_ * H_ + T_ * T_ + T_ * H_) * 2; // 294912 B
  (void)hipFuncSetAttribute((const void*)k_attn,
                            hipFuncAttributeMaxDynamicSharedMemorySize, smem);
  k_attn<<<B_ * NV_, 256, smem, stream>>>(node, neigh, bvec, G, WvT, outP, outA);
}
